// ESGGAT2Conv_12369505813021
// MI455X (gfx1250) — matrix operand order repaired and checked by reading, not compiled, not run on hardware
//
#include <hip/hip_runtime.h>
#include <hip/hip_fp16.h>
#include <math.h>

#define NN 100000
#define EE 600000
#define DD 128
#define HH 4
#define CC 32
#define AA 32
#define LL 2

typedef __attribute__((ext_vector_type(16))) _Float16 v16h;
typedef __attribute__((ext_vector_type(8)))  _Float16 v8h;
typedef __attribute__((ext_vector_type(4)))  _Float16 v4h;
typedef __attribute__((ext_vector_type(8)))  float    v8f;

// ---------------------------------------------------------------------------
// Atomic helpers
// ---------------------------------------------------------------------------
__device__ inline void atomAddF(float* p, float v) {
  unsafeAtomicAdd(p, v);  // lowers to global_atomic_add_f32
}

// Exact float atomic max via int/uint ordering trick (valid for mixed signs).
__device__ inline void atomMaxF(float* p, float v) {
  if (v >= 0.0f) atomicMax((int*)p, __float_as_int(v));
  else           atomicMin((unsigned int*)p, __float_as_uint(v));
}

// ---------------------------------------------------------------------------
// Pack weight W [Krows x 128] (row-major, f32) into WMMA B-fragment order:
// Wp[(kt*8+nt)*512 + lane*16 + i] = W[kt*32 + (i>>3)*16 + (lane>>4)*8 + (i&7)][nt*16 + (lane&15)]
// ---------------------------------------------------------------------------
__global__ void pack_b_k(const float* __restrict__ W, _Float16* __restrict__ Wp,
                         int Krows) {
  int tid = blockIdx.x * blockDim.x + threadIdx.x;
  int total = Krows * 128;
  if (tid >= total) return;
  int t    = tid >> 9;          // tile index (512 f16 per 32x16 tile)
  int lane = (tid >> 4) & 31;
  int i    = tid & 15;
  int kt = t >> 3, nt = t & 7;
  int row = kt * 32 + ((i >> 3) << 4) + ((lane >> 4) << 3) + (i & 7);
  int col = nt * 16 + (lane & 15);
  Wp[tid] = (_Float16)W[row * 128 + col];
}

__global__ void cast_f32_to_f16_k(const float* __restrict__ s,
                                  _Float16* __restrict__ d, long long n4) {
  long long t = (long long)blockIdx.x * blockDim.x + threadIdx.x;
  if (t >= n4) return;
  float4 v = ((const float4*)s)[t];
  v4h o;
  o[0] = (_Float16)v.x; o[1] = (_Float16)v.y;
  o[2] = (_Float16)v.z; o[3] = (_Float16)v.w;
  ((v4h*)d)[t] = o;
}

__global__ void fill_f32_k(float* __restrict__ p, float v, int n) {
  int t = blockIdx.x * blockDim.x + threadIdx.x;
  if (t < n) p[t] = v;
}

__global__ void init_out_k(float* __restrict__ out, const float* __restrict__ bias,
                           int n) {
  int t = blockIdx.x * blockDim.x + threadIdx.x;
  if (t < n) out[t] = bias[t & 127];
}

// ---------------------------------------------------------------------------
// WMMA GEMM: C[M,128] = A[M,K] @ B[K,128] (+bias). M multiple of 16, K = 32*ktiles.
// One wave per 16-row tile; A fragment reused across all 8 column tiles.
// Output f32 (Cf) or f16 (Ch).
// ---------------------------------------------------------------------------
__global__ __launch_bounds__(128) void gemm_wmma(
    const _Float16* __restrict__ A, int lda, int ktiles,
    const _Float16* __restrict__ Bp, const float* __restrict__ bias,
    float* __restrict__ Cf, _Float16* __restrict__ Ch, int mtiles)
{
  int tm = blockIdx.x * 4 + (threadIdx.x >> 5);
  if (tm >= mtiles) return;
  int l    = threadIdx.x & 31;
  int mrow = l & 15;
  int half = l >> 4;

  const _Float16* arow = A + (size_t)(tm * 16 + mrow) * lda + half * 8;

  v8f acc[8] = {};
  for (int kt = 0; kt < ktiles; ++kt) {
    // 16-bit A fragment: elems 0..7 = K base..base+7, elems 8..15 = K base+16..+23
    v8h alo = *(const v8h*)(arow + kt * 32);
    v8h ahi = *(const v8h*)(arow + kt * 32 + 16);
    v16h a = __builtin_shufflevector(alo, ahi,
                                     0,1,2,3,4,5,6,7,8,9,10,11,12,13,14,15);
    const _Float16* bbase = Bp + ((size_t)kt * 8 * 32 + l) * 16;
#pragma unroll
    for (int nt = 0; nt < 8; ++nt) {
      v16h b = *(const v16h*)(bbase + (size_t)nt * 512);
      acc[nt] = __builtin_amdgcn_wmma_f32_16x16x32_f16(
          false, a, false, b, (short)0, acc[nt], false, false);
    }
  }

  // C layout: lane l holds col nt*16+(l&15); VGPR r holds row r + 8*(l>>4)
  size_t rowbase = (size_t)(tm * 16 + half * 8) * 128;
#pragma unroll
  for (int nt = 0; nt < 8; ++nt) {
    int col = nt * 16 + mrow;
    float bv = bias ? bias[col] : 0.0f;
#pragma unroll
    for (int r = 0; r < 8; ++r) {
      float v = acc[nt][r] + bv;
      size_t idx = rowbase + (size_t)r * 128 + col;
      if (Cf) Cf[idx] = v;
      else    Ch[idx] = (_Float16)v;
    }
  }
}

// ---------------------------------------------------------------------------
// GATv2 attention logits: per edge, all 4 heads.
// logits[e,h] = sum_c lrelu(xl[src,h,c] + xr[dst,h,c] + ea[e,h,c]) * att[h,c]
// Also atomic-max into mx[dst,h].
// ---------------------------------------------------------------------------
__global__ __launch_bounds__(256) void logits_k(
    const float* __restrict__ xl, const float* __restrict__ xr,
    const _Float16* __restrict__ eah, const int* __restrict__ src,
    const int* __restrict__ dst, const float* __restrict__ att,
    float* __restrict__ logits, float* __restrict__ mx)
{
  __shared__ float s_att[128];
  if (threadIdx.x < 128) s_att[threadIdx.x] = att[threadIdx.x];
  __syncthreads();
  int e = blockIdx.x * 256 + threadIdx.x;
  if (e >= EE) return;
  int sn = src[e], tn = dst[e];
  const float4* pl = (const float4*)(xl + (size_t)sn * 128);
  const float4* pr = (const float4*)(xr + (size_t)tn * 128);
  const v4h*    pe = (const v4h*)(eah + (size_t)e * 128);
  float lg[4] = {0.f, 0.f, 0.f, 0.f};
#pragma unroll
  for (int q = 0; q < 32; ++q) {
    float4 a = pl[q];
    float4 b = pr[q];
    v4h    ev = pe[q];
    int h = q >> 3;
    float m;
    m = a.x + b.x + (float)ev[0]; m = m > 0.f ? m : 0.2f * m; lg[h] += m * s_att[q*4+0];
    m = a.y + b.y + (float)ev[1]; m = m > 0.f ? m : 0.2f * m; lg[h] += m * s_att[q*4+1];
    m = a.z + b.z + (float)ev[2]; m = m > 0.f ? m : 0.2f * m; lg[h] += m * s_att[q*4+2];
    m = a.w + b.w + (float)ev[3]; m = m > 0.f ? m : 0.2f * m; lg[h] += m * s_att[q*4+3];
  }
  ((float4*)logits)[e] = make_float4(lg[0], lg[1], lg[2], lg[3]);
#pragma unroll
  for (int h = 0; h < 4; ++h) atomMaxF(&mx[(size_t)tn * 4 + h], lg[h]);
}

// ex = exp(logit - mx[dst]); denom[dst] += ex  (overwrites logits with ex)
__global__ void expsum_k(float* __restrict__ logits, const float* __restrict__ mx,
                         const int* __restrict__ dst, float* __restrict__ denom) {
  int t = blockIdx.x * blockDim.x + threadIdx.x;
  if (t >= EE * HH) return;
  int e = t >> 2, h = t & 3;
  int tn = dst[e];
  float ex = expf(logits[t] - mx[(size_t)tn * 4 + h]);
  logits[t] = ex;
  atomAddF(&denom[(size_t)tn * 4 + h], ex);
}

// out[dst, :] += xl[src, :] * (ex / denom[dst])  — 4 channels per thread
__global__ __launch_bounds__(256) void scatter_k(
    const float* __restrict__ xl, const float* __restrict__ ex,
    const float* __restrict__ denom, const int* __restrict__ src,
    const int* __restrict__ dst, float* __restrict__ out)
{
  long long t = (long long)blockIdx.x * 256 + threadIdx.x;
  if (t >= (long long)EE * 32) return;
  int e  = (int)(t >> 5);
  int c4 = (int)(t & 31);
  int h  = c4 >> 3;
  int sn = src[e], tn = dst[e];
  float alpha = ex[(size_t)e * 4 + h] / denom[(size_t)tn * 4 + h];
  float4 v = ((const float4*)(xl + (size_t)sn * 128))[c4];
  float* o = out + (size_t)tn * 128 + c4 * 4;
  atomAddF(o + 0, v.x * alpha);
  atomAddF(o + 1, v.y * alpha);
  atomAddF(o + 2, v.z * alpha);
  atomAddF(o + 3, v.w * alpha);
}

// LayerNorm + exact GELU + residual, one wave (32 lanes x float4) per node row
__global__ __launch_bounds__(256) void ln_gelu_res_k(
    const float* __restrict__ h1, const float* __restrict__ gamma,
    const float* __restrict__ beta, float* __restrict__ xc)
{
  int wid  = (blockIdx.x * 256 + threadIdx.x) >> 5;
  int lane = threadIdx.x & 31;
  if (wid >= NN) return;
  float4 v = ((const float4*)(h1 + (size_t)wid * 128))[lane];
  float s  = v.x + v.y + v.z + v.w;
  float sq = v.x*v.x + v.y*v.y + v.z*v.z + v.w*v.w;
#pragma unroll
  for (int o = 16; o > 0; o >>= 1) {
    s  += __shfl_xor(s,  o, 32);
    sq += __shfl_xor(sq, o, 32);
  }
  float mu  = s * (1.0f / 128.0f);
  float var = sq * (1.0f / 128.0f) - mu * mu;
  float inv = rsqrtf(var + 1e-5f);
  float4 g  = ((const float4*)gamma)[lane];
  float4 bt = ((const float4*)beta)[lane];
  float4 xo = ((float4*)(xc + (size_t)wid * 128))[lane];
  float y;
  y = (v.x - mu)*inv*g.x + bt.x; xo.x += 0.5f*y*(1.0f + erff(y*0.70710678118f));
  y = (v.y - mu)*inv*g.y + bt.y; xo.y += 0.5f*y*(1.0f + erff(y*0.70710678118f));
  y = (v.z - mu)*inv*g.z + bt.z; xo.z += 0.5f*y*(1.0f + erff(y*0.70710678118f));
  y = (v.w - mu)*inv*g.w + bt.w; xo.w += 0.5f*y*(1.0f + erff(y*0.70710678118f));
  ((float4*)(xc + (size_t)wid * 128))[lane] = xo;
}

// ---------------------------------------------------------------------------
extern "C" void kernel_launch(void* const* d_in, const int* in_sizes, int n_in,
                              void* d_out, int out_size, void* d_ws, size_t ws_size,
                              hipStream_t stream)
{
  const float* x     = (const float*)d_in[0];
  const int*   eidx  = (const int*)d_in[1];
  const float* eattr = (const float*)d_in[2];
  const float* W_l   = (const float*)d_in[3];
  const float* b_l   = (const float*)d_in[4];
  const float* W_r   = (const float*)d_in[5];
  const float* b_r   = (const float*)d_in[6];
  const float* W_e   = (const float*)d_in[7];
  const float* att   = (const float*)d_in[8];
  const float* b_out = (const float*)d_in[9];
  const float* lin_W = (const float*)d_in[10];
  const float* lin_b = (const float*)d_in[11];
  const float* ln_g  = (const float*)d_in[12];
  const float* ln_bt = (const float*)d_in[13];
  const int* src = eidx;
  const int* dst = eidx + EE;

  char* ws = (char*)d_ws;
  float*    x_cur = (float*)(ws + 0ll);               //  51.2 MB
  _Float16* x16   = (_Float16*)(ws + 51200000ll);     //  25.6 MB (also out16)
  _Float16* ea16  = (_Float16*)(ws + 76800000ll);     //  38.4 MB
  float*    xl    = (float*)(ws + 115200000ll);       //  51.2 MB (also MLP h1)
  float*    xr    = (float*)(ws + 166400000ll);       //  51.2 MB
  _Float16* eah   = (_Float16*)(ws + 217600000ll);    // 153.6 MB
  float*    lgts  = (float*)(ws + 371200000ll);       //   9.6 MB
  float*    mx    = (float*)(ws + 380800000ll);       //   1.6 MB
  float*    den   = (float*)(ws + 382400000ll);       //   1.6 MB
  float*    outac = (float*)(ws + 384000000ll);       //  51.2 MB
  _Float16* Wl_pk = (_Float16*)(ws + 435200000ll);
  _Float16* Wr_pk = Wl_pk + 128 * 128;
  _Float16* ln_pk = Wr_pk + 128 * 128;
  _Float16* We_pk = ln_pk + 128 * 128;

  hipMemcpyAsync(x_cur, x, (size_t)NN * DD * 4, hipMemcpyDeviceToDevice, stream);
  {
    long long n4 = (long long)EE * AA / 4;
    cast_f32_to_f16_k<<<(n4 + 255) / 256, 256, 0, stream>>>(eattr, ea16, n4);
  }

  const int mtn = NN / 16;   // 6250
  const int mte = EE / 16;   // 37500

  for (int i = 0; i < LL; ++i) {
    pack_b_k<<<(128*128 + 255) / 256, 256, 0, stream>>>(W_l  + (size_t)i*DD*DD, Wl_pk, 128);
    pack_b_k<<<(128*128 + 255) / 256, 256, 0, stream>>>(W_r  + (size_t)i*DD*DD, Wr_pk, 128);
    pack_b_k<<<( 32*128 + 255) / 256, 256, 0, stream>>>(W_e  + (size_t)i*AA*DD, We_pk, 32);
    pack_b_k<<<(128*128 + 255) / 256, 256, 0, stream>>>(lin_W + (size_t)i*DD*DD, ln_pk, 128);

    {
      long long n4 = (long long)NN * DD / 4;
      cast_f32_to_f16_k<<<(n4 + 255) / 256, 256, 0, stream>>>(x_cur, x16, n4);
    }

    gemm_wmma<<<(mtn + 3) / 4, 128, 0, stream>>>(x16, 128, 4, Wl_pk, b_l + i*DD, xl, nullptr, mtn);
    gemm_wmma<<<(mtn + 3) / 4, 128, 0, stream>>>(x16, 128, 4, Wr_pk, b_r + i*DD, xr, nullptr, mtn);
    gemm_wmma<<<(mte + 3) / 4, 128, 0, stream>>>(ea16, 32, 1, We_pk, nullptr, nullptr, eah, mte);

    fill_f32_k<<<(NN*HH + 255) / 256, 256, 0, stream>>>(mx, -INFINITY, NN*HH);
    fill_f32_k<<<(NN*HH + 255) / 256, 256, 0, stream>>>(den, 0.0f, NN*HH);
    init_out_k<<<(NN*DD + 255) / 256, 256, 0, stream>>>(outac, b_out + i*DD, NN*DD);

    logits_k<<<(EE + 255) / 256, 256, 0, stream>>>(xl, xr, eah, src, dst,
                                                   att + i*HH*CC, lgts, mx);
    expsum_k<<<(EE*HH + 255) / 256, 256, 0, stream>>>(lgts, mx, dst, den);
    {
      long long n = (long long)EE * 32;
      scatter_k<<<(n + 255) / 256, 256, 0, stream>>>(xl, lgts, den, src, dst, outac);
    }

    {
      long long n4 = (long long)NN * DD / 4;
      cast_f32_to_f16_k<<<(n4 + 255) / 256, 256, 0, stream>>>(outac, x16, n4);
    }
    gemm_wmma<<<(mtn + 3) / 4, 128, 0, stream>>>(x16, 128, 4, ln_pk, lin_b + i*DD, xl, nullptr, mtn);
    ln_gelu_res_k<<<(NN + 7) / 8, 256, 0, stream>>>(xl, ln_g + i*DD, ln_bt + i*DD, x_cur);
  }

  hipMemcpyAsync(d_out, x_cur, (size_t)NN * DD * 4, hipMemcpyDeviceToDevice, stream);
}
